// Attention_MB_SSL_50594714747365
// MI455X (gfx1250) — matrix operand order repaired and checked by reading, not compiled, not run on hardware
//
#include <hip/hip_runtime.h>
#include <hip/hip_bf16.h>

// Sizes from the reference
#define NSEG 64
#define LDIM 1024
#define DDIM 256
#define FDIM 32

typedef _Float16 v16h __attribute__((ext_vector_type(16)));
typedef float    v8f  __attribute__((ext_vector_type(8)));
typedef float    v4f  __attribute__((ext_vector_type(4)));   // native vector: OK for nontemporal builtin

// ---------------------------------------------------------------------------
// Kernel 0: pre-swizzle W1 [1024 x 256] f32 -> f16 WMMA B-fragments.
// CDNA5 16-bit B matrix (32x16, KxN) layout per lane:
//   col  = lane & 15
//   K    = (lane>=16 ? 8 : 0) + h for h<8, 16 + (lane>=16 ? 8 : 0) + (h-8) for h>=8
// Fragment storage: BF[((kt*16 + ct)*32 + lane)*16 + h]
// ---------------------------------------------------------------------------
__global__ void k_prep_bfrag(const float* __restrict__ W1, _Float16* __restrict__ bf) {
  int tile = blockIdx.x;            // kt*16 + ct, 32*16 = 512 tiles
  int kt = tile >> 4, ct = tile & 15;
  int lane = threadIdx.x;           // 0..31
  int col = ct * 16 + (lane & 15);
  int kbase = kt * 32 + ((lane >> 4) * 8);
  _Float16* dst = bf + ((size_t)tile * 32 + lane) * 16;
#pragma unroll
  for (int h = 0; h < 16; ++h)
    dst[h] = (_Float16)W1[(size_t)(kbase + h + (h >= 8 ? 8 : 0)) * DDIM + col];
}

// ---------------------------------------------------------------------------
// Kernel 1: H = x @ W1 + b1 via v_wmma_f32_16x16x32_f16, software pipelined:
//   - all 8 B fragments (16 x b128 from L2) issued as one clause before WMMAs
//   - next K-step's A operand (x, non-temporal, 4 x b128) loaded before WMMAs
//   - global_prefetch on the x stream ahead of the loads
// Block = 256 threads = 8 waves; block tile = 64 rows x 256 cols.
// Wave w: row-tile rt = w&3 (16 rows), col-group cg = w>>2 (8 x 16 cols).
// CDNA5 16-bit A matrix (16x32) layout per lane:
//   row = lane & 15 ; kg = (lane>=16)?8:0
//   halves 0..7  -> K = kg + h ; halves 8..15 -> K = 16 + kg + (h-8)
// ---------------------------------------------------------------------------
__global__ void k_gemm(const float* __restrict__ x, const _Float16* __restrict__ bf,
                       const float* __restrict__ b1, float* __restrict__ H) {
  const int wave = threadIdx.x >> 5;
  const int lane = threadIdx.x & 31;
  const int rt = wave & 3;
  const int cg = wave >> 2;
  const size_t rowbase = (size_t)blockIdx.x * 64 + (size_t)rt * 16;
  const int mrow = lane & 15;
  const int kg4 = (lane >> 4) * 2;          // float4 index: kg/4 = 0 or 2
  const float* xrow = x + (rowbase + mrow) * LDIM;
  const v16h* bfv = (const v16h*)bf;

  v8f acc[8] = {};

  // Prologue: load A floats for kt = 0 (non-temporal: x is streamed once)
  const v4f* xp0 = (const v4f*)xrow;
  v4f p0 = __builtin_nontemporal_load(xp0 + kg4);
  v4f p1 = __builtin_nontemporal_load(xp0 + kg4 + 1);
  v4f p2 = __builtin_nontemporal_load(xp0 + kg4 + 4);
  v4f p3 = __builtin_nontemporal_load(xp0 + kg4 + 5);

  for (int kt = 0; kt < 32; ++kt) {
    // Issue all 8 B fragments (one clause of 16 b128) before any WMMA waits.
    v16h bb[8];
#pragma unroll
    for (int i = 0; i < 8; ++i)
      bb[i] = bfv[(size_t)(kt * 16 + cg * 8 + i) * 32 + lane];

    // Keep the HBM stream ahead of us.
    if (kt + 8 < 32)
      __builtin_prefetch(xrow + (kt + 8) * 32, 0, 0);

    // Load next K-step's A floats so HBM latency overlaps the WMMA block.
    v4f q0 = {}, q1 = {}, q2 = {}, q3 = {};
    if (kt + 1 < 32) {
      const v4f* xp = (const v4f*)(xrow + (kt + 1) * 32);
      q0 = __builtin_nontemporal_load(xp + kg4);
      q1 = __builtin_nontemporal_load(xp + kg4 + 1);
      q2 = __builtin_nontemporal_load(xp + kg4 + 4);
      q3 = __builtin_nontemporal_load(xp + kg4 + 5);
    }

    // Convert current A tile f32 -> f16 (packed cvt) into WMMA A layout.
    v16h a;
    a[0]  = (_Float16)p0.x; a[1]  = (_Float16)p0.y; a[2]  = (_Float16)p0.z; a[3]  = (_Float16)p0.w;
    a[4]  = (_Float16)p1.x; a[5]  = (_Float16)p1.y; a[6]  = (_Float16)p1.z; a[7]  = (_Float16)p1.w;
    a[8]  = (_Float16)p2.x; a[9]  = (_Float16)p2.y; a[10] = (_Float16)p2.z; a[11] = (_Float16)p2.w;
    a[12] = (_Float16)p3.x; a[13] = (_Float16)p3.y; a[14] = (_Float16)p3.z; a[15] = (_Float16)p3.w;

#pragma unroll
    for (int i = 0; i < 8; ++i)
      acc[i] = __builtin_amdgcn_wmma_f32_16x16x32_f16(
          /*neg_a=*/false, a, /*neg_b=*/false, bb[i],
          /*c_mod=*/(short)0, acc[i], /*reuse_a=*/false, /*reuse_b=*/false);

    p0 = q0; p1 = q1; p2 = q2; p3 = q3;
  }

  // C/D layout: VGPR v -> row = v + 8*(lane>=16), col = lane & 15 within tile
  const int chalf = (lane >> 4) * 8;
#pragma unroll
  for (int i = 0; i < 8; ++i) {
    int col = (cg * 8 + i) * 16 + (lane & 15);
    float bias = b1[col];
#pragma unroll
    for (int v = 0; v < 8; ++v) {
      size_t row = rowbase + (size_t)(v + chalf);
      H[row * DDIM + col] = acc[i][v] + bias;
    }
  }
}

// ---------------------------------------------------------------------------
// Kernel 2: logits = tanh(H @ Wa1 + ba1) @ Wa2 + ba2
// 4 lanes per row (lanes differ in low 2 bits -> same wave32), each covers
// 64 of the 256 D entries; cross-lane reduce with shfl_xor.
// ---------------------------------------------------------------------------
__global__ void k_logits(const float* __restrict__ H, const float* __restrict__ Wa1,
                         const float* __restrict__ ba1, const float* __restrict__ Wa2,
                         const float* __restrict__ ba2, float* __restrict__ logits) {
  const int t = threadIdx.x;
  const size_t row = (size_t)blockIdx.x * 64 + (size_t)(t >> 2);
  const int part = t & 3;
  const float* h = H + row * DDIM + part * 64;

  float pf[FDIM] = {};
  for (int d = 0; d < 64; ++d) {
    float hv = h[d];
    const float* wr = Wa1 + (size_t)(part * 64 + d) * FDIM;
#pragma unroll
    for (int f = 0; f < FDIM; ++f) pf[f] += hv * wr[f];
  }

  float logit = 0.f;
#pragma unroll
  for (int f = 0; f < FDIM; ++f) {
    float s = pf[f];
    s += __shfl_xor(s, 1, 32);
    s += __shfl_xor(s, 2, 32);
    if (part == 0) logit += tanhf(s + ba1[f]) * Wa2[f];
  }
  if (part == 0) logits[row] = logit + ba2[0];
}

// ---------------------------------------------------------------------------
// Kernel 3: per-bag max and sum(exp). idxs are sorted -> binary-search range.
// ---------------------------------------------------------------------------
__device__ __forceinline__ int lower_bound_i(const int* a, int n, int v) {
  int lo = 0, hi = n;
  while (lo < hi) { int m = (lo + hi) >> 1; if (a[m] < v) lo = m + 1; else hi = m; }
  return lo;
}

__global__ void k_segstats(const int* __restrict__ idx, const float* __restrict__ logits,
                           int N, float* __restrict__ segMax, float* __restrict__ segSum) {
  __shared__ float red[256];
  const int b = blockIdx.x, t = threadIdx.x;
  const int start = lower_bound_i(idx, N, b);
  const int end   = lower_bound_i(idx, N, b + 1);

  float mx = -3.402823466e38f;
  for (int i = start + t; i < end; i += 256) mx = fmaxf(mx, logits[i]);
  red[t] = mx; __syncthreads();
  for (int s = 128; s > 0; s >>= 1) { if (t < s) red[t] = fmaxf(red[t], red[t + s]); __syncthreads(); }
  const float m = red[0]; __syncthreads();

  float sum = 0.f;
  for (int i = start + t; i < end; i += 256) sum += expf(logits[i] - m);
  red[t] = sum; __syncthreads();
  for (int s = 128; s > 0; s >>= 1) { if (t < s) red[t] += red[t + s]; __syncthreads(); }
  if (t == 0) { segMax[b] = m; segSum[b] = red[0]; }
}

// Kernel 4: per-instance softmax weight
__global__ void k_weights(const int* __restrict__ idx, const float* __restrict__ logits,
                          const float* __restrict__ segMax, const float* __restrict__ segSum,
                          float* __restrict__ w, int N) {
  int i = blockIdx.x * 256 + threadIdx.x;
  if (i < N) { int s = idx[i]; w[i] = expf(logits[i] - segMax[s]) / segSum[s]; }
}

// Kernel 5: zero the output (d_out is poisoned before timing)
__global__ void k_zero(float* __restrict__ p, int n) {
  int i = blockIdx.x * 256 + threadIdx.x;
  if (i < n) p[i] = 0.f;
}

// ---------------------------------------------------------------------------
// Kernel 6: M[seg] += w_i * H_i. Thread = column; register accumulation over
// 256 sorted rows, atomicAdd only on segment transitions.
// ---------------------------------------------------------------------------
__global__ void k_segsumM(const int* __restrict__ idx, const float* __restrict__ w,
                          const float* __restrict__ H, float* __restrict__ M, int N) {
  const int t = threadIdx.x;  // column 0..255
  const size_t base = (size_t)blockIdx.x * 256;
  if ((long long)base >= N) return;
  const int rmax = (N - (int)base < 256) ? (N - (int)base) : 256;

  int cur = idx[base];
  float acc = 0.f;
  for (int r = 0; r < rmax; ++r) {
    size_t row = base + (size_t)r;
    int s = idx[row];
    if (s != cur) { atomicAdd(&M[(size_t)cur * DDIM + t], acc); acc = 0.f; cur = s; }
    acc += w[row] * H[row * DDIM + t];
  }
  atomicAdd(&M[(size_t)cur * DDIM + t], acc);
}

// ---------------------------------------------------------------------------
// Kernel 7: proj = normalize(M @ Wp + bp), torch F.normalize semantics.
// ---------------------------------------------------------------------------
__global__ void k_proj(const float* __restrict__ M, const float* __restrict__ Wp,
                       const float* __restrict__ bp, float* __restrict__ proj) {
  __shared__ float P[NSEG][FDIM + 1];
  const int t = threadIdx.x;
#pragma unroll
  for (int k = 0; k < 8; ++k) {
    int e = t * 8 + k;
    int r = e >> 5, f = e & 31;
    float s = bp[f];
    for (int d = 0; d < DDIM; ++d) s += M[(size_t)r * DDIM + d] * Wp[(size_t)d * FDIM + f];
    P[r][f] = s;
  }
  __syncthreads();
  if (t < NSEG) {
    float n2 = 0.f;
    for (int f = 0; f < FDIM; ++f) { float v = P[t][f]; n2 += v * v; }
    float inv = 1.0f / fmaxf(sqrtf(n2), 1e-12f);
    for (int f = 0; f < FDIM; ++f) proj[(size_t)t * FDIM + f] = P[t][f] * inv;
  }
}

// ---------------------------------------------------------------------------
extern "C" void kernel_launch(void* const* d_in, const int* in_sizes, int n_in,
                              void* d_out, int out_size, void* d_ws, size_t ws_size,
                              hipStream_t stream) {
  const float* x   = (const float*)d_in[0];
  const int*   idx = (const int*)  d_in[1];
  const float* W1  = (const float*)d_in[2];
  const float* b1  = (const float*)d_in[3];
  const float* Wa1 = (const float*)d_in[4];
  const float* ba1 = (const float*)d_in[5];
  const float* Wa2 = (const float*)d_in[6];
  const float* ba2 = (const float*)d_in[7];
  const float* Wp  = (const float*)d_in[8];
  const float* bp  = (const float*)d_in[9];

  const int N = in_sizes[1];  // number of instances

  // Workspace carve-up (all offsets 256B aligned)
  char* ws = (char*)d_ws;
  size_t o = 0;
  _Float16* BF = (_Float16*)(ws + o); o += (size_t)512 * 32 * 16 * sizeof(_Float16); // 512 KB
  float* H      = (float*)(ws + o); o += (size_t)N * DDIM * sizeof(float);           // 256 MiB
  float* logits = (float*)(ws + o); o += (size_t)N * sizeof(float);
  float* wgt    = (float*)(ws + o); o += (size_t)N * sizeof(float);
  float* segMax = (float*)(ws + o); o += 256;
  float* segSum = (float*)(ws + o); o += 256;

  float* outM = (float*)d_out;             // [64, 256]
  float* outP = outM + NSEG * DDIM;        // [64, 32]
  const int out_elems = NSEG * DDIM + NSEG * FDIM;

  k_prep_bfrag<<<512, 32, 0, stream>>>(W1, BF);
  k_gemm<<<N / 64, 256, 0, stream>>>(x, BF, b1, H);
  k_logits<<<N / 64, 256, 0, stream>>>(H, Wa1, ba1, Wa2, ba2, logits);
  k_segstats<<<NSEG, 256, 0, stream>>>(idx, logits, N, segMax, segSum);
  k_weights<<<(N + 255) / 256, 256, 0, stream>>>(idx, logits, segMax, segSum, wgt, N);
  k_zero<<<(out_elems + 255) / 256, 256, 0, stream>>>(outM, out_elems);
  k_segsumM<<<(N + 255) / 256, 256, 0, stream>>>(idx, wgt, H, outM, N);
  k_proj<<<1, 256, 0, stream>>>(outM, Wp, bp, outP);
}
